// Transformer_55662776156804
// MI455X (gfx1250) — matrix operand order repaired and checked by reading, not compiled, not run on hardware
//
#include <hip/hip_runtime.h>
#include <hip/hip_bf16.h>

typedef __attribute__((ext_vector_type(16))) _Float16 v16h;
typedef __attribute__((ext_vector_type(8)))  _Float16 v8h;
typedef __attribute__((ext_vector_type(8)))  float    v8f;

#define BB 64
#define TT 512
#define DMODEL 512
#define HH 8
#define DHD 64
#define DFF 2048
#define NEGVAL (-4294967295.0f)
#define SROW 516

// GEMM tiling
#define BLK_M 128
#define BLK_N 64
#define BLK_K 32
#define LDA 40   // padded row stride (halves): 80B rows -> conflict-free b128 LDS reads
#define LDB 40

// ---------------------------------------------------------------------------
// CDNA5 async global->LDS copy (16B per lane), tracked by ASYNCcnt.
// Inline asm (portable across ROCm7.2 / amdgpu-toolchain builtin arities).
// Generic LDS pointer low 32 bits == LDS byte offset; global generic == vaddr.
// ---------------------------------------------------------------------------
__device__ __forceinline__ void async_copy16(_Float16* lds_dst, const _Float16* gsrc) {
    unsigned           l = (unsigned)(uintptr_t)lds_dst;
    unsigned long long g = (unsigned long long)(uintptr_t)gsrc;
    asm volatile("global_load_async_to_lds_b128 %0, %1, off"
                 :: "v"(l), "v"(g) : "memory");
}
__device__ __forceinline__ void wait_async0() {
    asm volatile("s_wait_asynccnt 0x0" ::: "memory");
}

// ---------------------------------------------------------------------------
// position encoding: [cos(t*invf[d]) d<256 | sin(t*invf[d-256])], invf=10000^(-i/256)
// ---------------------------------------------------------------------------
__device__ __forceinline__ float pe_val(int t, int d) {
    int i = (d < DMODEL / 2) ? d : (d - DMODEL / 2);
    float inv = __powf(10000.0f, -(float)i / (float)(DMODEL / 2));
    float x = (float)t * inv;
    return (d < DMODEL / 2) ? __cosf(x) : __sinf(x);
}

__global__ void add_pe_kernel(const float* __restrict__ q, const float* __restrict__ k,
                              _Float16* __restrict__ qh, _Float16* __restrict__ kh) {
    size_t i = (size_t)blockIdx.x * blockDim.x + threadIdx.x;
    int d = (int)(i % DMODEL);
    int t = (int)((i / DMODEL) % TT);
    float pe = pe_val(t, d);
    qh[i] = (_Float16)(q[i] + pe);
    kh[i] = (_Float16)(k[i] + pe);
}

// W (K x N f32) -> Wt (N x K f16): WMMA B-fragments become contiguous loads
__global__ void transpose_w_kernel(const float* __restrict__ W, _Float16* __restrict__ Wt,
                                   int K, int N) {
    size_t i = (size_t)blockIdx.x * blockDim.x + threadIdx.x;   // over N*K
    int kk = (int)(i % K);
    int n  = (int)(i / K);
    Wt[i] = (_Float16)W[(size_t)kk * N + n];
}

// ---------------------------------------------------------------------------
// LDS-tiled, double-buffered, async-staged WMMA GEMM.
// C(MxN) = A(MxK f16 row-major) * Bt(NxK f16)^T
// Block: 256 thr (8 waves) -> 128x64 tile; wave(wm 0..3, wn 0..1) -> 32x32.
// ---------------------------------------------------------------------------
enum { OUT_F16_ROW = 0, OUT_VPT = 1, OUT_F32_RES = 2 };

template<int OUTMODE, bool RELU>
__global__ __launch_bounds__(256)
void gemm_nt_kernel(const _Float16* __restrict__ A, const _Float16* __restrict__ Bt,
                    void* __restrict__ Cout, const float* __restrict__ Res,
                    int M, int N, int K) {
    __shared__ _Float16 As[2 * BLK_M * LDA];   // 20.0 KB
    __shared__ _Float16 Bs[2 * BLK_N * LDB];   // 10.0 KB

    const int tid  = (int)threadIdx.x;
    const int lane = tid & 31;
    const int lr = lane & 15, lh = lane >> 4;
    const int wave = tid >> 5;
    const int wm = wave & 3;        // 0..3 : row quarter
    const int wn = wave >> 2;       // 0..1 : col half

    const int nTiles = N / BLK_N;
    const int m0 = ((int)blockIdx.x / nTiles) * BLK_M;
    const int n0 = ((int)blockIdx.x % nTiles) * BLK_N;

    // async stage of one k-slice into buffer `buf`
    auto stage = [&](int buf, int k0) {
        _Float16* as = As + buf * BLK_M * LDA;
        _Float16* bs = Bs + buf * BLK_N * LDB;
#pragma unroll
        for (int p = 0; p < 2; p++) {          // A: 128 rows x 4 chunks = 512
            int q = tid + p * 256;
            int row = q >> 2, ch = q & 3;
            async_copy16(as + row * LDA + ch * 8,
                         A + (size_t)(m0 + row) * K + k0 + ch * 8);
        }
        {                                      // B: 64 rows x 4 chunks = 256
            int row = tid >> 2, ch = tid & 3;
            async_copy16(bs + row * LDB + ch * 8,
                         Bt + (size_t)(n0 + row) * K + k0 + ch * 8);
        }
    };

    v8f acc[2][2] = {};
    const int KT = K / BLK_K;
    stage(0, 0);
    for (int kt = 0; kt < KT; kt++) {
        wait_async0();
        __syncthreads();
        if (kt + 1 < KT) stage((kt + 1) & 1, (kt + 1) * BLK_K);

        const _Float16* as = As + (kt & 1) * BLK_M * LDA;
        const _Float16* bs = Bs + (kt & 1) * BLK_N * LDB;

        v16h af[2], bf[2];
#pragma unroll
        for (int mi = 0; mi < 2; mi++) {
            // A 16x32 fragment: elems 0..7 -> K=lh*8+0..7, 8..15 -> K=16+lh*8+0..7
            const _Float16* ap = as + (wm * 32 + mi * 16 + lr) * LDA + lh * 8;
            v8h a0 = *(const v8h*)ap;
            v8h a1 = *(const v8h*)(ap + 16);
            af[mi] = __builtin_shufflevector(a0, a1, 0,1,2,3,4,5,6,7,8,9,10,11,12,13,14,15);
        }
#pragma unroll
        for (int ni = 0; ni < 2; ni++) {
            // B 32x16 fragment: lane owns column, elems 0..7 -> K=lh*8+0..7, 8..15 -> K=16+lh*8+0..7
            const _Float16* bp = bs + (wn * 32 + ni * 16 + lr) * LDB + lh * 8;
            v8h b0 = *(const v8h*)bp;
            v8h b1 = *(const v8h*)(bp + 16);
            bf[ni] = __builtin_shufflevector(b0, b1, 0,1,2,3,4,5,6,7,8,9,10,11,12,13,14,15);
        }
#pragma unroll
        for (int mi = 0; mi < 2; mi++)
#pragma unroll
            for (int ni = 0; ni < 2; ni++)
                acc[mi][ni] = __builtin_amdgcn_wmma_f32_16x16x32_f16(
                    false, af[mi], false, bf[ni], (short)0, acc[mi][ni], false, false);
        __syncthreads();
    }

#pragma unroll
    for (int mi = 0; mi < 2; mi++) {
#pragma unroll
        for (int ni = 0; ni < 2; ni++) {
            int nn = n0 + wn * 32 + ni * 16 + lr;
#pragma unroll
            for (int j = 0; j < 8; j++) {
                int mm = m0 + wm * 32 + mi * 16 + lh * 8 + j;   // C/D: M = lh*8 + vgpr
                float v = acc[mi][ni][j];
                if (RELU) v = v > 0.0f ? v : 0.0f;
                if (OUTMODE == OUT_F16_ROW) {
                    ((_Float16*)Cout)[(size_t)mm * N + nn] = (_Float16)v;
                } else if (OUTMODE == OUT_VPT) {
                    // vp transposed per head: vpT[b][h][d][t]
                    int b = mm / TT, t = mm % TT;
                    int h = nn >> 6, d = nn & 63;
                    ((_Float16*)Cout)[(((size_t)(b * HH + h) * DHD + d) * TT) + t] = (_Float16)v;
                } else {
                    ((float*)Cout)[(size_t)mm * N + nn] = v + Res[(size_t)mm * N + nn];
                }
            }
        }
    }
}

// ---------------------------------------------------------------------------
// Fused attention: one wave per (b, h, 16-query tile).
// ---------------------------------------------------------------------------
__global__ void attn_kernel(const _Float16* __restrict__ qp, const _Float16* __restrict__ kp,
                            const _Float16* __restrict__ vpT, const int* __restrict__ qlen,
                            const int* __restrict__ klen, float* __restrict__ out) {
    __shared__ float sc[16 * SROW];
    int blk = (int)blockIdx.x;               // B*H*(T/16)
    int tTile = blk & 31;
    int h = (blk >> 5) & 7;
    int b = blk >> 8;
    int lane = (int)threadIdx.x & 31;
    int lr = lane & 15, lh = lane >> 4;
    int kl = klen[b], ql = qlen[b];

    const _Float16* qrow = qp + ((size_t)(b * TT + (tTile << 4) + lr)) * DMODEL + h * DHD + lh * 16;
    v16h aq0, aq1;
#pragma unroll
    for (int i = 0; i < 8; i++) {
        int kk = i << 1;
        aq0[2 * i] = qrow[kk];      aq0[2 * i + 1] = qrow[kk + 1];
        aq1[2 * i] = qrow[32 + kk]; aq1[2 * i + 1] = qrow[32 + kk + 1];
    }

    for (int st = 0; st < 32; st++) {
        int s = (st << 4) + lr;
        const _Float16* krow = kp + ((size_t)(b * TT + s)) * DMODEL + h * DHD + lh * 16;
        v16h bf0 = *(const v16h*)(krow);
        v16h bf1 = *(const v16h*)(krow + 32);
        v8f accs = {};
        accs = __builtin_amdgcn_wmma_f32_16x16x32_f16(false, aq0, false, bf0, (short)0, accs, false, false);
        accs = __builtin_amdgcn_wmma_f32_16x16x32_f16(false, aq1, false, bf1, (short)0, accs, false, false);
#pragma unroll
        for (int j = 0; j < 8; j++) {
            float v = accs[j] * 0.125f;      // 1/sqrt(64)
            if (s >= kl) v = NEGVAL;
            sc[(lh * 8 + j) * SROW + s] = v;
        }
    }
    __syncthreads();

    if (lane < 16) {
        float* row = sc + lane * SROW;
        float mx = -3.4e38f;
        for (int c = 0; c < TT; c++) mx = fmaxf(mx, row[c]);
        float sum = 0.0f;
        for (int c = 0; c < TT; c++) { float e = __expf(row[c] - mx); sum += e; row[c] = e; }
        float scale = ((tTile << 4) + lane < ql) ? (1.0f / sum) : 0.0f;   // q_mask
        for (int c = 0; c < TT; c++) row[c] *= scale;
    }
    __syncthreads();

    v8f acc[4] = {};
    for (int st = 0; st < 16; st++) {
        v16h af;
        const float* prow = sc + lr * SROW + (st << 5) + lh * 16;
#pragma unroll
        for (int i = 0; i < 8; i++) {
            int kk = i << 1;
            af[2 * i]     = (_Float16)prow[kk];
            af[2 * i + 1] = (_Float16)prow[kk + 1];
        }
#pragma unroll
        for (int nt = 0; nt < 4; nt++) {
            const _Float16* vrow = vpT + (((size_t)(b * HH + h) * DHD) + (nt << 4) + lr) * TT
                                       + (st << 5) + lh * 16;
            v16h bf = *(const v16h*)vrow;
            acc[nt] = __builtin_amdgcn_wmma_f32_16x16x32_f16(
                false, af, false, bf, (short)0, acc[nt], false, false);
        }
    }
#pragma unroll
    for (int nt = 0; nt < 4; nt++) {
#pragma unroll
        for (int j = 0; j < 8; j++) {
            int t = (tTile << 4) + lh * 8 + j;
            int d = h * DHD + (nt << 4) + lr;
            out[((size_t)(b * TT + t)) * DMODEL + d] = acc[nt][j];
        }
    }
}

__global__ void residual_kernel(const float* __restrict__ q, float* __restrict__ resid,
                                _Float16* __restrict__ residh) {
    size_t i = (size_t)blockIdx.x * blockDim.x + threadIdx.x;
    int d = (int)(i % DMODEL);
    int t = (int)((i / DMODEL) % TT);
    float v = resid[i] + q[i] + pe_val(t, d);
    resid[i] = v;
    residh[i] = (_Float16)v;
}

__global__ void mean_kernel(const float* __restrict__ fin, float* __restrict__ out) {
    int i = (int)(blockIdx.x * blockDim.x + threadIdx.x);   // B*D
    int d = i % DMODEL;
    int b = i / DMODEL;
    float s = 0.0f;
    const float* p = fin + (size_t)b * TT * DMODEL + d;
    for (int t = 0; t < TT; t++) s += p[(size_t)t * DMODEL];
    out[i] = s * (1.0f / TT);
}

extern "C" void kernel_launch(void* const* d_in, const int* in_sizes, int n_in,
                              void* d_out, int out_size, void* d_ws, size_t ws_size,
                              hipStream_t stream) {
    (void)in_sizes; (void)n_in; (void)out_size; (void)ws_size;
    const float* queries = (const float*)d_in[0];
    const float* keys    = (const float*)d_in[1];
    const int*   qlen    = (const int*)d_in[2];
    const int*   klen    = (const int*)d_in[3];
    const float* W_Q = (const float*)d_in[4];
    const float* W_K = (const float*)d_in[5];
    const float* W_V = (const float*)d_in[6];
    const float* fw1 = (const float*)d_in[7];
    const float* fw2 = (const float*)d_in[8];

    char* ws = (char*)d_ws;
    size_t off = 0;
    auto alloc = [&](size_t bytes) -> char* {
        char* p = ws + off;
        off += (bytes + 255) & ~(size_t)255;
        return p;
    };
    const size_t BTD = (size_t)BB * TT * DMODEL;
    _Float16* qin_h  = (_Float16*)alloc(BTD * 2);
    _Float16* kin_h  = (_Float16*)alloc(BTD * 2);
    _Float16* wq_t   = (_Float16*)alloc((size_t)DMODEL * DMODEL * 2);
    _Float16* wk_t   = (_Float16*)alloc((size_t)DMODEL * DMODEL * 2);
    _Float16* wv_t   = (_Float16*)alloc((size_t)DMODEL * DMODEL * 2);
    _Float16* fw1_t  = (_Float16*)alloc((size_t)DMODEL * DFF * 2);
    _Float16* fw2_t  = (_Float16*)alloc((size_t)DFF * DMODEL * 2);
    _Float16* qp_h   = (_Float16*)alloc(BTD * 2);
    _Float16* kp_h   = (_Float16*)alloc(BTD * 2);
    _Float16* vpT_h  = (_Float16*)alloc(BTD * 2);
    float*    resid  = (float*)alloc(BTD * 4);
    _Float16* residh = (_Float16*)alloc(BTD * 2);
    _Float16* h1_h   = (_Float16*)alloc((size_t)BB * TT * DFF * 2);
    float*    fin    = (float*)alloc(BTD * 4);

    const int M = BB * TT;   // 32768

    // 1) PE add + f16 convert
    add_pe_kernel<<<(unsigned)(BTD / 256), 256, 0, stream>>>(queries, keys, qin_h, kin_h);

    // 2) weight transposes (f32 KxN -> f16 NxK)
    transpose_w_kernel<<<(DMODEL * DMODEL) / 256, 256, 0, stream>>>(W_Q, wq_t, DMODEL, DMODEL);
    transpose_w_kernel<<<(DMODEL * DMODEL) / 256, 256, 0, stream>>>(W_K, wk_t, DMODEL, DMODEL);
    transpose_w_kernel<<<(DMODEL * DMODEL) / 256, 256, 0, stream>>>(W_V, wv_t, DMODEL, DMODEL);
    transpose_w_kernel<<<(DMODEL * DFF) / 256, 256, 0, stream>>>(fw1, fw1_t, DMODEL, DFF);
    transpose_w_kernel<<<(DFF * DMODEL) / 256, 256, 0, stream>>>(fw2, fw2_t, DFF, DMODEL);

    // 3) projections (async-staged WMMA GEMMs)
    unsigned gProj = (unsigned)((M / BLK_M) * (DMODEL / BLK_N));
    gemm_nt_kernel<OUT_F16_ROW, false><<<gProj, 256, 0, stream>>>(qin_h, wq_t, qp_h, nullptr, M, DMODEL, DMODEL);
    gemm_nt_kernel<OUT_F16_ROW, false><<<gProj, 256, 0, stream>>>(kin_h, wk_t, kp_h, nullptr, M, DMODEL, DMODEL);
    // DeepCTR quirk: V projected from projected keys; stored per-head transposed
    gemm_nt_kernel<OUT_VPT, false><<<gProj, 256, 0, stream>>>(kp_h, wv_t, vpT_h, nullptr, M, DMODEL, DMODEL);

    // 4) fused masked attention (writes attention output into resid buffer)
    attn_kernel<<<BB * HH * (TT / 16), 32, 0, stream>>>(qp_h, kp_h, vpT_h, qlen, klen, resid);

    // 5) residual with PE-augmented queries
    residual_kernel<<<(unsigned)(BTD / 256), 256, 0, stream>>>(queries, resid, residh);

    // 6) FFN: relu(resid @ fw1) @ fw2 + resid
    unsigned gFF1 = (unsigned)((M / BLK_M) * (DFF / BLK_N));
    gemm_nt_kernel<OUT_F16_ROW, true><<<gFF1, 256, 0, stream>>>(residh, fw1_t, h1_h, nullptr, M, DFF, DMODEL);
    gemm_nt_kernel<OUT_F32_RES, false><<<gProj, 256, 0, stream>>>(h1_h, fw2_t, fin, resid, M, DMODEL, DFF);

    // 7) mean over T
    mean_kernel<<<(BB * DMODEL) / 256, 256, 0, stream>>>(fin, (float*)d_out);
}
